// ConstellationLayer_21285857919540
// MI455X (gfx1250) — compile-verified, matrix operand order checked by reading, not run on hardware
//
#include <hip/hip_runtime.h>
#include <hip/hip_bf16.h>
#include <math.h>

// dims
#define Bn   32
#define Cn   512
#define HWn  1024
#define PWn  256
#define En   256
#define An   32
#define NCn  8
#define DCn  32
#define Rn   24
#define Kn   3
#define NSUB 10   // subspace iterations
#define JSW  10   // jacobi sweeps

typedef __bf16 bf16;
typedef __attribute__((ext_vector_type(16))) __bf16 v16bf;
typedef __attribute__((ext_vector_type(8)))  float  v8f;

// ---------------------------------------------------------------- helpers
__device__ __forceinline__ float wave_sum(float v) {
#pragma unroll
  for (int o = 16; o; o >>= 1) v += __shfl_xor(v, o, 32);
  return v;
}

__device__ __forceinline__ float block_reduce_sum(float v, float* red) {
  const int t = threadIdx.x;
  red[t] = v;
  __syncthreads();
#pragma unroll
  for (int s = 128; s > 0; s >>= 1) {
    if (t < s) red[t] += red[t + s];
    __syncthreads();
  }
  float r = red[0];
  __syncthreads();
  return r;
}

__device__ __forceinline__ float gelu_exact(float x) {
  return 0.5f * x * (1.0f + erff(x * 0.70710678118654752f));
}
__device__ __forceinline__ float sigmoidf(float x) { return 1.0f / (1.0f + expf(-x)); }

// ------------------------------------------------- K1: Gram via bf16 WMMA
// 16-bit A-matrix 16x32 fragment layout (ISA 7.12.2): lane<16 -> m=lane,
// K pairs {0..7} in v0..3 and {16..23} in v4..7; lane>=16 -> K {8..15},{24..31}.
__device__ __forceinline__ v16bf load_frag(const bf16* panel, int rowBase, int lane) {
  v16bf f;
  const int mrow = rowBase + (lane & 15);
  const int half = lane >> 4;
  const bf16* pr = panel + mrow * 32;
#pragma unroll
  for (int v = 0; v < 8; ++v) {
    const int k = ((v & 3) << 1) + ((v >> 2) << 4) + (half << 3);
    f[2 * v]     = pr[k];
    f[2 * v + 1] = pr[k + 1];
  }
  return f;
}

// Block tile: 128 (M) x 64 (N); 8 waves; each wave owns 4 subtiles in one
// 16-wide column strip -> one B-fragment pair feeds 4 accumulators
// (12 v_wmma per wave per k-step). LDS: A 128x32 hi/lo + B 64x32 hi/lo = 24KB.
__global__ __launch_bounds__(256) void gram_wmma_kernel(const float* __restrict__ x,
                                                        float* __restrict__ gram) {
  __shared__ bf16 Ah[128 * 32], Al[128 * 32], Bh[64 * 32], Bl[64 * 32];
  const int t = threadIdx.x, lane = t & 31, wave = t >> 5;
  const int b = blockIdx.z, tr = blockIdx.y, tc = blockIdx.x;
  const int gr0 = tr * 128, gc0 = tc * 64;
  const float* xb = x + (size_t)b * Cn * HWn;
  // A panel: thread loads 16 contiguous floats of one row (128 rows x 32 k)
  const int rrA = t >> 1, ccA = (t & 1) * 16;
  // B panel: thread loads 8 contiguous floats (64 rows x 32 k)
  const int rrB = t >> 2, ccB = (t & 3) * 8;
  const int sr = wave >> 2;        // 0..1: row subtile phase
  const int n0 = (wave & 3) * 16;  // column subtile

  v8f acc[4];
#pragma unroll
  for (int i = 0; i < 4; ++i) acc[i] = (v8f){0.f, 0.f, 0.f, 0.f, 0.f, 0.f, 0.f, 0.f};

  for (int k0 = 0; k0 < HWn; k0 += 32) {
    const float* ga = xb + (size_t)(gr0 + rrA) * HWn + k0 + ccA;
    const float* gb = xb + (size_t)(gc0 + rrB) * HWn + k0 + ccB;
    if (k0 + 32 < HWn) {  // pull next k-panel into WGP-level caches
      __builtin_prefetch(ga + 32, 0, 3);
      __builtin_prefetch(gb + 32, 0, 3);
    }
#pragma unroll
    for (int i = 0; i < 16; ++i) {
      float va = ga[i];
      bf16 ha = (bf16)va;
      Ah[rrA * 32 + ccA + i] = ha;
      Al[rrA * 32 + ccA + i] = (bf16)(va - (float)ha);
    }
#pragma unroll
    for (int i = 0; i < 8; ++i) {
      float vb = gb[i];
      bf16 hb = (bf16)vb;
      Bh[rrB * 32 + ccB + i] = hb;
      Bl[rrB * 32 + ccB + i] = (bf16)(vb - (float)hb);
    }
    __syncthreads();

    const v16bf bh = load_frag(Bh, n0, lane);
    const v16bf bl = load_frag(Bl, n0, lane);
#pragma unroll
    for (int i = 0; i < 4; ++i) {
      const int rbase = sr * 16 + i * 32;
      const v16bf ah = load_frag(Ah, rbase, lane);
      const v16bf al = load_frag(Al, rbase, lane);
      acc[i] = __builtin_amdgcn_wmma_f32_16x16x32_bf16(false, ah, false, bh, (short)0, acc[i], false, false);
      acc[i] = __builtin_amdgcn_wmma_f32_16x16x32_bf16(false, ah, false, bl, (short)0, acc[i], false, false);
      acc[i] = __builtin_amdgcn_wmma_f32_16x16x32_bf16(false, al, false, bh, (short)0, acc[i], false, false);
    }
    __syncthreads();
  }

  // C/D layout: lanes 0-15: M=v, N=lane; lanes 16-31: M=v+8, N=lane-16
  const float invHW = 1.0f / (float)HWn;
  const int half = lane >> 4, n = lane & 15;
  float* g0 = gram + (size_t)b * Cn * Cn;
#pragma unroll
  for (int i = 0; i < 4; ++i) {
#pragma unroll
    for (int v = 0; v < 8; ++v) {
      const int m = v + half * 8;
      g0[(size_t)(gr0 + sr * 16 + i * 32 + m) * Cn + gc0 + n0 + n] = acc[i][v] * invHW;
    }
  }
}

// ------------------------------------------------- K2: per-(b,c) spatial mean
__global__ __launch_bounds__(256) void colmean_kernel(const float* __restrict__ x,
                                                      float* __restrict__ xm) {
  const int lane = threadIdx.x & 31;
  const int waveId = blockIdx.x * 8 + (threadIdx.x >> 5); // 0..16383 == b*512+c
  const float* p = x + (size_t)waveId * HWn;
  float s = 0.f;
#pragma unroll
  for (int i = 0; i < 32; ++i) s += p[lane + 32 * i];
  s = wave_sum(s);
  if (lane == 0) xm[waveId] = s * (1.0f / (float)HWn);
}

// -------------------------------------- K3: subspace iteration eigensolver
__device__ void matmul_GV(const float* __restrict__ G, const float* __restrict__ V,
                          float* __restrict__ Y) {
  for (int idx = threadIdx.x; idx < Cn * 32; idx += 256) {
    const int r = idx >> 5, j = idx & 31;
    const float* gr = G + (size_t)r * Cn;
    float s = 0.f;
#pragma unroll 8
    for (int k = 0; k < Cn; ++k) s += gr[k] * V[k * 32 + j];
    Y[idx] = s;
  }
}

__global__ __launch_bounds__(256) void subspace_eig_kernel(const float* __restrict__ gram,
                                                           float* __restrict__ Vg,
                                                           float* __restrict__ Yg,
                                                           float* __restrict__ svd_raw) {
  __shared__ float red[256];
  __shared__ float Ts[32 * 32];
  __shared__ float dots[32];
  __shared__ float ccs[16], sss[16];
  __shared__ int   ppa[16], qqa[16];

  const int t = threadIdx.x;
  const int b = blockIdx.x;
  const float* G = gram + (size_t)b * Cn * Cn;
  float* V = Vg + (size_t)b * Cn * 32;
  float* Y = Yg + (size_t)b * Cn * 32;

  // deterministic pseudo-random init of V
  for (int idx = t; idx < Cn * 32; idx += 256) {
    unsigned h = (unsigned)idx * 2654435761u ^ ((unsigned)b * 1013904223u + 0x9E3779B9u);
    h ^= h >> 16; h *= 0x85EBCA6Bu; h ^= h >> 13;
    V[idx] = (float)(h >> 8) * (1.0f / 8388608.0f) - 1.0f;
  }
  __syncthreads();

  for (int it = 0; it < NSUB; ++it) {
    matmul_GV(G, V, Y);
    __syncthreads();
    // modified Gram-Schmidt on the 32 columns of Y (each 512 long)
    for (int j = 0; j < 32; ++j) {
      float p = Y[t * 32 + j] * Y[t * 32 + j] +
                Y[(t + 256) * 32 + j] * Y[(t + 256) * 32 + j];
      float n2 = block_reduce_sum(p, red);
      float inv = rsqrtf(fmaxf(n2, 1e-24f));
      Y[t * 32 + j] *= inv;
      Y[(t + 256) * 32 + j] *= inv;
      __syncthreads();
      // dot(col j, col j2) for all j2 simultaneously
      const int g = t >> 5, j2 = t & 31;
      float d = 0.f;
#pragma unroll 8
      for (int i = 0; i < 64; ++i) {
        const int k = g + (i << 3);
        d += Y[k * 32 + j] * Y[k * 32 + j2];
      }
      red[t] = d;
      __syncthreads();
      if (t < 32) {
        float s = 0.f;
#pragma unroll
        for (int gg = 0; gg < 8; ++gg) s += red[gg * 32 + t];
        dots[t] = s;
      }
      __syncthreads();
      for (int idx = t; idx < Cn * 32; idx += 256) {
        const int r = idx >> 5, jj = idx & 31;
        if (jj > j) Y[idx] -= dots[jj] * Y[r * 32 + j];
      }
      __syncthreads();
    }
    // V = Y
    for (int idx = t; idx < Cn * 32; idx += 256) V[idx] = Y[idx];
    __syncthreads();
  }

  // Rayleigh-Ritz: T = V^T (G V)
  matmul_GV(G, V, Y);
  __syncthreads();
  for (int idx = t; idx < 1024; idx += 256) {
    const int j = idx >> 5, k2 = idx & 31;
    float s = 0.f;
#pragma unroll 8
    for (int r = 0; r < Cn; ++r) s += V[r * 32 + j] * Y[r * 32 + k2];
    Ts[idx] = s;
  }
  __syncthreads();
  // symmetrize (each (j<k) pair owned by exactly one thread)
  for (int idx = t; idx < 1024; idx += 256) {
    const int j = idx >> 5, k2 = idx & 31;
    if (j < k2) {
      float a = 0.5f * (Ts[j * 32 + k2] + Ts[k2 * 32 + j]);
      Ts[j * 32 + k2] = a;
      Ts[k2 * 32 + j] = a;
    }
  }
  __syncthreads();

  // parallel cyclic Jacobi (round-robin tournament, 16 disjoint pairs/round)
  for (int sw = 0; sw < JSW; ++sw) {
    for (int r = 0; r < 31; ++r) {
      if (t < 16) {
        int p, q;
        if (t == 0) { p = 31; q = r % 31; }
        else        { p = (r + t) % 31; q = (r + 31 - t) % 31; }
        if (p > q) { int tmp = p; p = q; q = tmp; }
        ppa[t] = p; qqa[t] = q;
        const float app = Ts[p * 32 + p], aqq = Ts[q * 32 + q], apq = Ts[p * 32 + q];
        float c, s;
        if (fabsf(apq) < 1e-20f) { c = 1.f; s = 0.f; }
        else {
          const float tau = (aqq - app) / (2.0f * apq);
          const float sg = (tau >= 0.f) ? 1.f : -1.f;
          const float tt = sg / (fabsf(tau) + sqrtf(1.0f + tau * tau));
          c = rsqrtf(1.0f + tt * tt);
          s = tt * c;
        }
        ccs[t] = c; sss[t] = s;
      }
      __syncthreads();
      { // left rotations (rows p,q) - rows disjoint across pairs
        const int i = t >> 4, col = t & 15;
        const int p = ppa[i], q = qqa[i];
        const float c = ccs[i], s = sss[i];
#pragma unroll
        for (int cc2 = col; cc2 < 32; cc2 += 16) {
          const float a = Ts[p * 32 + cc2], bb = Ts[q * 32 + cc2];
          Ts[p * 32 + cc2] = c * a - s * bb;
          Ts[q * 32 + cc2] = s * a + c * bb;
        }
      }
      __syncthreads();
      { // right rotations (cols p,q)
        const int i = t >> 4, row = t & 15;
        const int p = ppa[i], q = qqa[i];
        const float c = ccs[i], s = sss[i];
#pragma unroll
        for (int rr2 = row; rr2 < 32; rr2 += 16) {
          const float a = Ts[rr2 * 32 + p], bb = Ts[rr2 * 32 + q];
          Ts[rr2 * 32 + p] = c * a - s * bb;
          Ts[rr2 * 32 + q] = s * a + c * bb;
        }
      }
      __syncthreads();
    }
  }

  // sort eigenvalues descending via rank, emit svd_raw = log1p(sqrt(clip(ev)))
  if (t < 32) {
    const float d = Ts[t * 32 + t];
    int rank = 0;
    for (int u = 0; u < 32; ++u) {
      const float du = Ts[u * 32 + u];
      rank += (du > d) || (du == d && u < t);
    }
    if (rank < Rn) svd_raw[b * Rn + rank] = log1pf(sqrtf(fmaxf(d, 0.f)));
  }
}

// ------------------------------------------------- K4: svd_ctx = gelu(LN(raw@W+b))
__global__ __launch_bounds__(256) void svdctx_kernel(const float* __restrict__ svd_raw,
                                                     const float* __restrict__ W_svd,
                                                     const float* __restrict__ b_svd,
                                                     const float* __restrict__ g_svd,
                                                     const float* __restrict__ beta_svd,
                                                     float* __restrict__ svd_ctx) {
  __shared__ float red[256];
  const int t = threadIdx.x, b = blockIdx.x;
  float s = b_svd[t];
#pragma unroll
  for (int r = 0; r < Rn; ++r) s += svd_raw[b * Rn + r] * W_svd[r * PWn + t];
  const float mean = block_reduce_sum(s, red) * (1.0f / PWn);
  const float d = s - mean;
  const float var = block_reduce_sum(d * d, red) * (1.0f / PWn);
  const float ln = d * rsqrtf(var + 1e-5f) * g_svd[t] + beta_svd[t];
  svd_ctx[b * PWn + t] = gelu_exact(ln);
}

// ------------------------------------------------- K5: embedding (LN + L2 norm)
__global__ __launch_bounds__(256) void emb_kernel(const float* __restrict__ xm,
                                                  const float* __restrict__ W_emb,
                                                  const float* __restrict__ g_emb,
                                                  const float* __restrict__ beta_emb,
                                                  float* __restrict__ embw) {
  __shared__ float red[256];
  const int t = threadIdx.x, b = blockIdx.x;
  const float* xb = xm + b * Cn;
  float s = 0.f;
#pragma unroll 8
  for (int c = 0; c < Cn; ++c) s += xb[c] * W_emb[c * En + t];
  const float mean = block_reduce_sum(s, red) * (1.0f / En);
  const float d = s - mean;
  const float var = block_reduce_sum(d * d, red) * (1.0f / En);
  const float ln = d * rsqrtf(var + 1e-5f) * g_emb[t] + beta_emb[t];
  const float n2 = block_reduce_sum(ln * ln, red);
  embw[b * En + t] = ln / fmaxf(sqrtf(n2), 1e-12f);
}

// ------------------- K6: anchor cosine + top-k gate + grouped compartment MLPs
__global__ __launch_bounds__(256) void gate_mlp_kernel(const float* __restrict__ embw,
                                                       const float* __restrict__ anchors,
                                                       const float* __restrict__ Wc1,
                                                       const float* __restrict__ bc1,
                                                       const float* __restrict__ Wc2,
                                                       const float* __restrict__ bc2,
                                                       const float* __restrict__ g_c,
                                                       const float* __restrict__ beta_c,
                                                       float* __restrict__ pww) {
  __shared__ float embL[En];
  __shared__ float cosL[An];
  __shared__ float trig[An];
  __shared__ float hsh[NCn * 2 * DCn];
  const int t = threadIdx.x, lane = t & 31, wave = t >> 5;
  const int b = blockIdx.x;

  embL[t] = embw[b * En + t];
  if (t < An) trig[t] = 0.f;
  __syncthreads();

  // cosines: each wave does 4 anchors (anchors normalized on the fly)
#pragma unroll
  for (int ii = 0; ii < 4; ++ii) {
    const int a = wave + ii * 8;
    float da = 0.f, na = 0.f;
#pragma unroll
    for (int i = 0; i < 8; ++i) {
      const int e = lane + 32 * i;
      const float av = anchors[a * En + e];
      da += embL[e] * av;
      na += av * av;
    }
    da = wave_sum(da);
    na = wave_sum(na);
    if (lane == 0) cosL[a] = da / fmaxf(sqrtf(na), 1e-12f);
  }
  __syncthreads();

  // top-3 + softmax gate (serial on thread 0; 32 entries)
  if (t == 0) {
    int   bi[Kn];
    float bv[Kn];
    bool used[An];
    for (int a = 0; a < An; ++a) used[a] = false;
    for (int k = 0; k < Kn; ++k) {
      int best = 0; float bb = -1e30f;
      for (int a = 0; a < An; ++a)
        if (!used[a] && cosL[a] > bb) { bb = cosL[a]; best = a; }
      used[best] = true; bi[k] = best; bv[k] = bb;
    }
    const float m = bv[0];
    float e0 = expf(bv[0] - m), e1 = expf(bv[1] - m), e2 = expf(bv[2] - m);
    const float inv = 1.0f / (e0 + e1 + e2);
    trig[bi[0]] = (1.0f - bv[0]) * e0 * inv;
    trig[bi[1]] = (1.0f - bv[1]) * e1 * inv;
    trig[bi[2]] = (1.0f - bv[2]) * e2 * inv;
  }
  __syncthreads();

  // h = gelu(trig @ Wc1[c] + bc1[c])  -- wave w == compartment w, lane does d, d+32
  {
    const int c = wave;
#pragma unroll
    for (int half = 0; half < 2; ++half) {
      const int d = lane + half * 32;
      float s = bc1[c * 64 + d];
#pragma unroll
      for (int a = 0; a < An; ++a) s += trig[a] * Wc1[(c * An + a) * 64 + d];
      hsh[c * 64 + d] = gelu_exact(s);
    }
  }
  __syncthreads();

  // pc = h @ Wc2[c] + bc2[c], then LN over DC=32 inside the wave
  {
    const int c = wave, e = lane;
    float s = bc2[c * DCn + e];
#pragma unroll
    for (int d = 0; d < 64; ++d) s += hsh[c * 64 + d] * Wc2[(c * 64 + d) * DCn + e];
    const float mean = wave_sum(s) * (1.0f / DCn);
    const float dd = s - mean;
    const float var = wave_sum(dd * dd) * (1.0f / DCn);
    const float ln = dd * rsqrtf(var + 1e-5f) * g_c[c * DCn + e] + beta_c[c * DCn + e];
    pww[b * PWn + c * DCn + e] = ln;
  }
}

// ------------------------------------------------- K7: channel modulation weights
__global__ __launch_bounds__(256) void mod_kernel(const float* __restrict__ pww,
                                                  const float* __restrict__ svd_ctx,
                                                  const float* __restrict__ geo_state,
                                                  const float* __restrict__ W_mod,
                                                  const float* __restrict__ b_mod,
                                                  float* __restrict__ cww) {
  const int t = threadIdx.x;
  const int cch = blockIdx.x * 256 + t;
  const int b = blockIdx.y;
  float s = b_mod[cch];
#pragma unroll 4
  for (int i = 0; i < PWn; ++i) s += pww[b * PWn + i] * W_mod[(size_t)i * Cn + cch];
#pragma unroll 4
  for (int i = 0; i < PWn; ++i) s += svd_ctx[b * PWn + i] * W_mod[(size_t)(PWn + i) * Cn + cch];
#pragma unroll 4
  for (int i = 0; i < PWn; ++i) s += geo_state[b * PWn + i] * W_mod[(size_t)(2 * PWn + i) * Cn + cch];
  cww[b * Cn + cch] = sigmoidf(s);
}

// ------------------------------------------------- K8: x_out = x * cw (float4)
__global__ __launch_bounds__(256) void xout_kernel(const float* __restrict__ x,
                                                   const float* __restrict__ cww,
                                                   float* __restrict__ out) {
  const int i4 = blockIdx.x * 256 + threadIdx.x; // 4,194,304 total
  const int i = i4 << 2;
  const int c = (i >> 10) & (Cn - 1);
  const int b = i >> 19; // 512*1024 = 2^19
  const float w = cww[b * Cn + c];
  const float4 xv = ((const float4*)x)[i4];
  float4 o;
  o.x = xv.x * w; o.y = xv.y * w; o.z = xv.z * w; o.w = xv.w * w;
  ((float4*)out)[i4] = o;
}

// ------------------------------------------------- K9: gated residual geo update
__global__ __launch_bounds__(256) void geo_kernel(const float* __restrict__ pww,
                                                  const float* __restrict__ W_geo,
                                                  const float* __restrict__ b_geo,
                                                  const float* __restrict__ g_geo,
                                                  const float* __restrict__ beta_geo,
                                                  const float* __restrict__ geo_gate,
                                                  const float* __restrict__ geo_state,
                                                  float* __restrict__ geo_out) {
  __shared__ float red[256];
  const int t = threadIdx.x, b = blockIdx.x;
  float s = b_geo[t];
#pragma unroll 4
  for (int i = 0; i < PWn; ++i) s += pww[b * PWn + i] * W_geo[i * PWn + t];
  const float mean = block_reduce_sum(s, red) * (1.0f / PWn);
  const float d = s - mean;
  const float var = block_reduce_sum(d * d, red) * (1.0f / PWn);
  const float ln = d * rsqrtf(var + 1e-5f) * g_geo[t] + beta_geo[t];
  const float g = sigmoidf(geo_gate[t]);
  geo_out[b * PWn + t] = geo_state[b * PWn + t] + g * ln;
}

// ---------------------------------------------------------------- launcher
extern "C" void kernel_launch(void* const* d_in, const int* in_sizes, int n_in,
                              void* d_out, int out_size, void* d_ws, size_t ws_size,
                              hipStream_t stream) {
  const float* x         = (const float*)d_in[0];
  const float* geo_state = (const float*)d_in[1];
  const float* W_svd     = (const float*)d_in[2];
  const float* b_svd     = (const float*)d_in[3];
  const float* g_svd     = (const float*)d_in[4];
  const float* beta_svd  = (const float*)d_in[5];
  const float* W_emb     = (const float*)d_in[6];
  const float* g_emb     = (const float*)d_in[7];
  const float* beta_emb  = (const float*)d_in[8];
  const float* anchors   = (const float*)d_in[9];
  const float* Wc1       = (const float*)d_in[10];
  const float* bc1       = (const float*)d_in[11];
  const float* Wc2       = (const float*)d_in[12];
  const float* bc2       = (const float*)d_in[13];
  const float* g_c       = (const float*)d_in[14];
  const float* beta_c    = (const float*)d_in[15];
  const float* W_mod     = (const float*)d_in[16];
  const float* b_mod     = (const float*)d_in[17];
  const float* geo_gate  = (const float*)d_in[18];
  const float* W_geo     = (const float*)d_in[19];
  const float* b_geo     = (const float*)d_in[20];
  const float* g_geo     = (const float*)d_in[21];
  const float* beta_geo  = (const float*)d_in[22];

  float* out_x   = (float*)d_out;                       // 16,777,216 floats
  float* out_geo = out_x + (size_t)Bn * Cn * HWn;       // 8,192 floats

  // workspace layout (floats)
  float* wsf     = (float*)d_ws;
  float* gram    = wsf;                                  // 32*512*512 = 8,388,608
  float* Vg      = gram + (size_t)Bn * Cn * Cn;          // 524,288
  float* Yg      = Vg + (size_t)Bn * Cn * 32;            // 524,288
  float* xm      = Yg + (size_t)Bn * Cn * 32;            // 16,384
  float* svd_raw = xm + Bn * Cn;                         // 768
  float* svd_ctx = svd_raw + Bn * Rn;                    // 8,192
  float* embw    = svd_ctx + Bn * PWn;                   // 8,192
  float* pww     = embw + Bn * En;                       // 8,192
  float* cww     = pww + Bn * PWn;                       // 16,384

  // K1: per-batch Gram via bf16 hi/lo WMMA (128x64 block tiles)
  gram_wmma_kernel<<<dim3(8, 4, Bn), 256, 0, stream>>>(x, gram);
  // K2: channel means of x
  colmean_kernel<<<dim3((Bn * Cn) / 8), 256, 0, stream>>>(x, xm);
  // K3: top-R eigenvalues via subspace iteration + Rayleigh-Ritz Jacobi
  subspace_eig_kernel<<<dim3(Bn), 256, 0, stream>>>(gram, Vg, Yg, svd_raw);
  // K4: svd context
  svdctx_kernel<<<dim3(Bn), 256, 0, stream>>>(svd_raw, W_svd, b_svd, g_svd, beta_svd, svd_ctx);
  // K5: embedding
  emb_kernel<<<dim3(Bn), 256, 0, stream>>>(xm, W_emb, g_emb, beta_emb, embw);
  // K6: cosine gate + compartment MLPs -> pw
  gate_mlp_kernel<<<dim3(Bn), 256, 0, stream>>>(embw, anchors, Wc1, bc1, Wc2, bc2, g_c, beta_c, pww);
  // K7: modulation weights
  mod_kernel<<<dim3(Cn / 256, Bn), 256, 0, stream>>>(pww, svd_ctx, geo_state, W_mod, b_mod, cww);
  // K8: x_out
  xout_kernel<<<dim3((Bn * Cn * HWn) / (4 * 256)), 256, 0, stream>>>(x, cww, out_x);
  // K9: geo_out
  geo_kernel<<<dim3(Bn), 256, 0, stream>>>(pww, W_geo, b_geo, g_geo, beta_geo, geo_gate, geo_state, out_geo);
}
